// TransformerNetV2_22978075034138
// MI455X (gfx1250) — compile-verified, matrix operand order checked by reading, not run on hardware
//
#include <hip/hip_runtime.h>

typedef __attribute__((ext_vector_type(16))) _Float16 v16h;
typedef __attribute__((ext_vector_type(8)))  _Float16 v8h;
typedef __attribute__((ext_vector_type(8)))  float    v8f;

#define B_   2048
#define S_   60
#define F_   20
#define D_   256
#define SA_  61
#define ML_  70
#define DFF_ 1024

static __device__ __forceinline__ float geluf(float x) {
  return 0.5f * x * (1.f + erff(x * 0.70710678118654752f));
}

static __device__ __forceinline__ float blockReduceSum(float v, float* red) {
  #pragma unroll
  for (int off = 16; off; off >>= 1) v += __shfl_xor(v, off, 32);
  const int wave = threadIdx.x >> 5, lane = threadIdx.x & 31;
  if (lane == 0) red[wave] = v;
  __syncthreads();
  float t = (wave == 0 && lane < 8) ? red[lane] : 0.f;
  if (wave == 0) {
    #pragma unroll
    for (int off = 4; off; off >>= 1) t += __shfl_xor(t, off, 32);
    if (lane == 0) red[0] = t;
  }
  __syncthreads();
  float r = red[0];
  __syncthreads();
  return r;
}

// Load one 16x32 f16 WMMA fragment (A layout; B uses the same pattern on W rows).
static __device__ __forceinline__ v16h ldfrag(const _Float16* __restrict__ row,
                                              int k, int kh) {
  v8h lo = *(const v8h*)(row + k + kh);
  v8h hi = *(const v8h*)(row + k + 16 + kh);
  v16h r;
  #pragma unroll
  for (int i = 0; i < 8; ++i) { r[i] = lo[i]; r[i + 8] = hi[i]; }
  return r;
}

// ---------------------------------------------------------------------------
// WMMA GEMM:  C[M,N] = A[M,K](f16,row-major) * W[N,K](f16,row-major)^T
// Block = 8 waves arranged 2(m) x 4(n); each wave owns a 32x32 C macro-tile
// (2x2 register tiling -> 4 independent WMMA accumulator chains per wave).
// Block tile = 64 x 128.  M % 64 == 0, N % 128 == 0, K % 32 == 0.
// mode 0: f32 out (+bias)  1: gelu -> f16  2: +resid -> f32  3: f16
// ---------------------------------------------------------------------------
__global__ void __launch_bounds__(256)
gemm_wmma(const _Float16* __restrict__ A, const _Float16* __restrict__ W,
          const float* __restrict__ bias, const float* __restrict__ resid,
          float* __restrict__ outF, _Float16* __restrict__ outH,
          int M, int N, int K, int mode)
{
  const int wave = threadIdx.x >> 5;
  const int lane = threadIdx.x & 31;
  const int mbase = blockIdx.x * 64 + (wave >> 2) * 32;
  const int nbase = blockIdx.y * 128 + (wave & 3) * 32;
  const int kh = (lane >> 4) * 8;            // K-half offset per lane group

  const _Float16* Ar0 = A + (size_t)(mbase + (lane & 15)) * K;
  const _Float16* Ar1 = Ar0 + (size_t)16 * K;
  const _Float16* Wr0 = W + (size_t)(nbase + (lane & 15)) * K;
  const _Float16* Wr1 = Wr0 + (size_t)16 * K;

  v8f acc[2][2] = {};
  for (int k = 0; k < K; k += 32) {
    if (k + 64 < K) {
      __builtin_prefetch(Ar0 + k + 64, 0, 1);
      __builtin_prefetch(Ar1 + k + 64, 0, 1);
      __builtin_prefetch(Wr0 + k + 64, 0, 1);
      __builtin_prefetch(Wr1 + k + 64, 0, 1);
    }
    const v16h a0 = ldfrag(Ar0, k, kh);
    const v16h a1 = ldfrag(Ar1, k, kh);
    const v16h b0 = ldfrag(Wr0, k, kh);
    const v16h b1 = ldfrag(Wr1, k, kh);
    acc[0][0] = __builtin_amdgcn_wmma_f32_16x16x32_f16(false, a0, false, b0,
                                                       (short)0, acc[0][0], false, false);
    acc[0][1] = __builtin_amdgcn_wmma_f32_16x16x32_f16(false, a0, false, b1,
                                                       (short)0, acc[0][1], false, false);
    acc[1][0] = __builtin_amdgcn_wmma_f32_16x16x32_f16(false, a1, false, b0,
                                                       (short)0, acc[1][0], false, false);
    acc[1][1] = __builtin_amdgcn_wmma_f32_16x16x32_f16(false, a1, false, b1,
                                                       (short)0, acc[1][1], false, false);
  }
  #pragma unroll
  for (int ti = 0; ti < 2; ++ti) {
    #pragma unroll
    for (int tj = 0; tj < 2; ++tj) {
      const int n  = nbase + tj * 16 + (lane & 15);
      const int mb = mbase + ti * 16 + (lane >> 4) * 8;
      const float bv = bias ? bias[n] : 0.f;
      #pragma unroll
      for (int r = 0; r < 8; ++r) {
        const size_t idx = (size_t)(mb + r) * N + n;
        float v = acc[ti][tj][r] + bv;
        if (mode == 2)      outF[idx] = v + resid[idx];
        else if (mode == 1) outH[idx] = (_Float16)geluf(v);
        else if (mode == 3) outH[idx] = (_Float16)v;
        else                outF[idx] = v;
      }
    }
  }
}

// f32 -> f16 conversion (weights)
__global__ void __launch_bounds__(256)
f32_to_f16(const float* __restrict__ in, _Float16* __restrict__ out, size_t n)
{
  size_t i = (size_t)blockIdx.x * 256 + threadIdx.x;
  if (i < n) out[i] = (_Float16)in[i];
}

// ---------------------------------------------------------------------------
// Feature embedding: GELU(LN(x @ emb_w^T + emb_b)) + Time2Vec.  1 block/token.
// ---------------------------------------------------------------------------
__global__ void __launch_bounds__(256)
embed_kernel(const float* __restrict__ x, const float* __restrict__ ew,
             const float* __restrict__ eb, const float* __restrict__ g,
             const float* __restrict__ be, const float* __restrict__ wl,
             const float* __restrict__ bl, const float* __restrict__ wp,
             const float* __restrict__ bp, float* __restrict__ h0)
{
  __shared__ float xs[F_];
  __shared__ float red[8];
  const int token = blockIdx.x;            // b*S + s
  const int s = token % S_;
  if (threadIdx.x < F_) xs[threadIdx.x] = x[(size_t)token * F_ + threadIdx.x];
  __syncthreads();
  const int d = threadIdx.x;
  float acc = eb[d];
  #pragma unroll
  for (int f = 0; f < F_; ++f) acc += ew[d * F_ + f] * xs[f];
  float mean = blockReduceSum(acc, red) * (1.f / D_);
  float df = acc - mean;
  float var = blockReduceSum(df * df, red) * (1.f / D_);
  float y = df * rsqrtf(var + 1e-5f) * g[d] + be[d];
  y = geluf(y);
  const float pos = (float)s;
  const float t2v = (d == 0) ? (wl[0] * pos + bl[0])
                             : sinf(wp[d - 1] * pos + bp[d - 1]);
  h0[(size_t)token * D_ + d] = y + t2v;
}

// ---------------------------------------------------------------------------
// Depthwise-separable conv branches (k=3,5,7) + GELU + residual + LN.
// 1 block/token; writes into residual stream at seq position t+1.
// ---------------------------------------------------------------------------
__global__ void __launch_bounds__(256)
conv_kernel(const float* __restrict__ h0,
            const float* __restrict__ dw0, const float* __restrict__ db0,
            const float* __restrict__ pw0, const float* __restrict__ pb0,
            const float* __restrict__ dw1, const float* __restrict__ db1,
            const float* __restrict__ pw1, const float* __restrict__ pb1,
            const float* __restrict__ dw2, const float* __restrict__ db2,
            const float* __restrict__ pw2, const float* __restrict__ pb2,
            const float* __restrict__ g, const float* __restrict__ be,
            float* __restrict__ hout)
{
  __shared__ float win[7 * D_];
  __shared__ float dws[3 * D_];
  __shared__ float red[8];
  const int token = blockIdx.x;
  const int b = token / S_, t = token % S_;
  for (int i = threadIdx.x; i < 7 * D_; i += 256) {
    const int j = i / D_, d = i % D_;
    const int sr = t + j - 3;
    win[i] = (sr >= 0 && sr < S_) ? h0[((size_t)b * S_ + sr) * D_ + d] : 0.f;
  }
  __syncthreads();
  const int d = threadIdx.x;
  float a0 = db0[d], a1 = db1[d], a2 = db2[d];
  #pragma unroll
  for (int j = 0; j < 3; ++j) a0 += dw0[d * 3 + j] * win[(j + 2) * D_ + d];
  #pragma unroll
  for (int j = 0; j < 5; ++j) a1 += dw1[d * 5 + j] * win[(j + 1) * D_ + d];
  #pragma unroll
  for (int j = 0; j < 7; ++j) a2 += dw2[d * 7 + j] * win[j * D_ + d];
  dws[d] = a0; dws[D_ + d] = a1; dws[2 * D_ + d] = a2;
  __syncthreads();
  const float* pw; const float* pb; int idx, br;
  if (d < 85)       { pw = pw0; pb = pb0; idx = d;       br = 0; }
  else if (d < 170) { pw = pw1; pb = pb1; idx = d - 85;  br = 1; }
  else              { pw = pw2; pb = pb2; idx = d - 170; br = 2; }
  float acc = pb[idx];
  const float* src = &dws[br * D_];
  for (int dd = 0; dd < D_; ++dd) acc += pw[idx * D_ + dd] * src[dd];
  acc = geluf(acc);
  const float r = win[3 * D_ + d] + acc;     // residual (row t itself)
  float mean = blockReduceSum(r, red) * (1.f / D_);
  float df = r - mean;
  float var = blockReduceSum(df * df, red) * (1.f / D_);
  float y = df * rsqrtf(var + 1e-5f) * g[d] + be[d];
  hout[((size_t)b * SA_ + (t + 1)) * D_ + d] = y;
}

__global__ void __launch_bounds__(256)
cls_kernel(const float* __restrict__ cls, float* __restrict__ h)
{
  h[(size_t)blockIdx.x * SA_ * D_ + threadIdx.x] = cls[threadIdx.x];
}

// LayerNorm over D=256; one block per row; writes f16 for WMMA consumption.
// rowStride in rows (61 picks CLS rows from the (B,61,D) stream).
__global__ void __launch_bounds__(256)
ln_kernel(const float* __restrict__ in, const float* __restrict__ g,
          const float* __restrict__ be, _Float16* __restrict__ out16,
          int rowStride)
{
  __shared__ float red[8];
  const size_t row = (size_t)blockIdx.x * rowStride;
  float v = in[row * D_ + threadIdx.x];
  float mean = blockReduceSum(v, red) * (1.f / D_);
  float df = v - mean;
  float var = blockReduceSum(df * df, red) * (1.f / D_);
  float y = df * rsqrtf(var + 1e-5f) * g[threadIdx.x] + be[threadIdx.x];
  out16[(size_t)blockIdx.x * D_ + threadIdx.x] = (_Float16)y;
}

// ---------------------------------------------------------------------------
// Attention core: 1 block per (b, head). q/k/v staged in LDS (k transposed),
// one wave per query row; optional relative-position bias table (H,139).
// Writes concatenated head outputs as f16 for the out-proj WMMA GEMM.
// ---------------------------------------------------------------------------
__global__ void __launch_bounds__(256)
attn_kernel(const float* __restrict__ qkv, const float* __restrict__ relT,
            _Float16* __restrict__ out, int nhead, int hd)
{
  extern __shared__ float smem[];
  const int bh = blockIdx.x;
  const int b = bh / nhead, h = bh % nhead;
  float* q_s = smem;
  float* k_s = q_s + SA_ * hd;
  float* v_s = k_s + SA_ * hd;
  float* p_s = v_s + SA_ * hd;               // 8 waves * 64
  const int tq = 3 * D_;
  const float* base = qkv + (size_t)b * SA_ * tq + h * hd;
  for (int i = threadIdx.x; i < SA_ * hd; i += 256) {
    const int s = i / hd, d = i % hd;
    q_s[s * hd + d]  = base[(size_t)s * tq + d];
    k_s[d * SA_ + s] = base[(size_t)s * tq + D_ + d];
    v_s[s * hd + d]  = base[(size_t)s * tq + 2 * D_ + d];
  }
  __syncthreads();
  const float scale = rsqrtf((float)hd);
  const int wave = threadIdx.x >> 5, lane = threadIdx.x & 31;
  const int k1 = lane + 32;
  float* pw = p_s + wave * 64;
  for (int qi = wave; qi < SA_; qi += 8) {
    float s0 = 0.f, s1 = 0.f;
    for (int d = 0; d < hd; ++d) {
      const float qv = q_s[qi * hd + d];
      s0 += qv * k_s[d * SA_ + lane];
      if (k1 < SA_) s1 += qv * k_s[d * SA_ + k1];
    }
    s0 *= scale; s1 *= scale;
    if (relT) {
      s0 += relT[h * (2 * ML_ - 1) + (qi - lane + ML_ - 1)];
      if (k1 < SA_) s1 += relT[h * (2 * ML_ - 1) + (qi - k1 + ML_ - 1)];
    }
    const float sv1 = (k1 < SA_) ? s1 : -3.0e38f;
    float m = fmaxf(s0, sv1);
    #pragma unroll
    for (int off = 16; off; off >>= 1) m = fmaxf(m, __shfl_xor(m, off, 32));
    const float e0 = expf(s0 - m);
    const float e1 = (k1 < SA_) ? expf(sv1 - m) : 0.f;
    float sum = e0 + e1;
    #pragma unroll
    for (int off = 16; off; off >>= 1) sum += __shfl_xor(sum, off, 32);
    const float inv = 1.f / sum;
    pw[lane] = e0 * inv;
    pw[lane + 32] = e1 * inv;
    for (int d = lane; d < hd; d += 32) {
      float acc = 0.f;
      for (int k2 = 0; k2 < SA_; ++k2) acc += pw[k2] * v_s[k2 * hd + d];
      out[((size_t)b * SA_ + qi) * D_ + h * hd + d] = (_Float16)acc;
    }
  }
}

// GLU: val * sigmoid(gate) over fc1 output (B,1024) -> f16 (B,512)
__global__ void __launch_bounds__(256)
glu_kernel(const float* __restrict__ in, _Float16* __restrict__ out)
{
  const int i = blockIdx.x * 256 + threadIdx.x;
  const int b = i >> 9, j = i & 511;
  const float val  = in[(size_t)b * 1024 + j];
  const float gate = in[(size_t)b * 1024 + 512 + j];
  out[i] = (_Float16)(val / (1.f + expf(-gate)));
}

// Final projection to scalar per batch row.
__global__ void __launch_bounds__(256)
proj_kernel(const float* __restrict__ in, const float* __restrict__ w,
            const float* __restrict__ b, float* __restrict__ out)
{
  __shared__ float red[8];
  const float v = in[(size_t)blockIdx.x * D_ + threadIdx.x] * w[threadIdx.x];
  const float s = blockReduceSum(v, red);
  if (threadIdx.x == 0) out[blockIdx.x] = s + b[0];
}

// ---------------------------------------------------------------------------
extern "C" void kernel_launch(void* const* d_in, const int* in_sizes, int n_in,
                              void* d_out, int out_size, void* d_ws, size_t ws_size,
                              hipStream_t stream)
{
  (void)in_sizes; (void)n_in; (void)out_size; (void)ws_size;
  auto IN = [&](int i) { return (const float*)d_in[i]; };
  auto AL = [](size_t x) { return (x + 255) & ~(size_t)255; };

  char* ws = (char*)d_ws;
  const size_t tok = (size_t)B_ * SA_;                       // 124928
  float*    hbuf = (float*)ws;        size_t off = AL(tok * D_ * 4);
  _Float16* a16  = (_Float16*)(ws + off); off += AL(tok * D_ * 2);
  float*    big  = (float*)(ws + off);    off += AL(tok * 3 * D_ * 4);
  _Float16* o16  = (_Float16*)(ws + off); off += AL(tok * D_ * 2);
  _Float16* w16  = (_Float16*)(ws + off);

  // Aliases (disjoint lifetimes on the single stream):
  float*    h0     = big;                       // embed out (B,S,D) f32
  _Float16* mid16  = (_Float16*)big;            // FFN mid (tok,1024) f16
  float*    fc1out = big;                       // head fc1 (B,1024) f32
  float*    fc2out = (float*)((char*)big + AL((size_t)B_ * DFF_ * 4));
  _Float16* glu16  = o16;                       // head GLU (B,512) f16
  _Float16* cls16  = a16;                       // head LN  (B,256) f16

  const size_t W_IN = 768 * 256, W_OUT = 256 * 256, W_W1 = 1024 * 256,
               W_W2 = 256 * 1024, W_L = W_IN + W_OUT + W_W1 + W_W2;
  _Float16 *wIn[4], *wOut[4], *wW1[4], *wW2[4];
  for (int l = 0; l < 4; ++l) {
    _Float16* p = w16 + (size_t)l * W_L;
    wIn[l] = p; wOut[l] = p + W_IN; wW1[l] = p + W_IN + W_OUT;
    wW2[l] = p + W_IN + W_OUT + W_W1;
  }
  _Float16* wFiIn  = w16 + 4 * W_L;
  _Float16* wFiOut = wFiIn + W_IN;
  _Float16* wFc1   = wFiOut + W_OUT;
  _Float16* wFc2   = wFc1 + 1024 * 256;

  auto cvt = [&](const float* s, _Float16* d, size_t n) {
    f32_to_f16<<<(unsigned)((n + 255) / 256), 256, 0, stream>>>(s, d, n);
  };
  for (int l = 0; l < 4; ++l) {
    const int base = 24 + 13 * l;
    cvt(IN(base + 0),  wIn[l],  W_IN);
    cvt(IN(base + 2),  wOut[l], W_OUT);
    cvt(IN(base + 9),  wW1[l],  W_W1);
    cvt(IN(base + 11), wW2[l],  W_W2);
  }
  cvt(IN(76), wFiIn, W_IN);
  cvt(IN(78), wFiOut, W_OUT);
  cvt(IN(84), wFc1, 1024 * 256);
  cvt(IN(86), wFc2, 256 * 512);

  // Embedding + Time2Vec, conv block, CLS prepend.
  embed_kernel<<<B_ * S_, 256, 0, stream>>>(IN(0), IN(1), IN(2), IN(3), IN(4),
                                            IN(5), IN(6), IN(7), IN(8), h0);
  conv_kernel<<<B_ * S_, 256, 0, stream>>>(h0,
      IN(9), IN(10), IN(11), IN(12), IN(13), IN(14), IN(15), IN(16),
      IN(17), IN(18), IN(19), IN(20), IN(21), IN(22), hbuf);
  cls_kernel<<<B_, 256, 0, stream>>>(IN(23), hbuf);

  const int M = (int)tok;                       // 124928, multiple of 64
  const unsigned MT = (unsigned)(M / 64);
  const size_t smA = (size_t)(3 * SA_ * 32 + 512) * 4;
  const size_t smF = (size_t)(3 * SA_ * 64 + 512) * 4;

  for (int l = 0; l < 4; ++l) {
    const int base = 24 + 13 * l;
    ln_kernel<<<M, 256, 0, stream>>>(hbuf, IN(base + 5), IN(base + 6), a16, 1);
    gemm_wmma<<<dim3(MT, 6), 256, 0, stream>>>(a16, wIn[l], IN(base + 1),
        nullptr, big, nullptr, M, 768, 256, 0);
    attn_kernel<<<B_ * 8, 256, smA, stream>>>(big, IN(base + 4), o16, 8, 32);
    gemm_wmma<<<dim3(MT, 2), 256, 0, stream>>>(o16, wOut[l], IN(base + 3),
        hbuf, hbuf, nullptr, M, 256, 256, 2);
    ln_kernel<<<M, 256, 0, stream>>>(hbuf, IN(base + 7), IN(base + 8), a16, 1);
    gemm_wmma<<<dim3(MT, 8), 256, 0, stream>>>(a16, wW1[l], IN(base + 10),
        nullptr, nullptr, mid16, M, 1024, 256, 1);
    gemm_wmma<<<dim3(MT, 2), 256, 0, stream>>>(mid16, wW2[l], IN(base + 12),
        hbuf, hbuf, nullptr, M, 256, 1024, 2);
  }

  // Feature-interaction MHA (4 heads, hd=64, no bias).
  ln_kernel<<<M, 256, 0, stream>>>(hbuf, IN(80), IN(81), a16, 1);
  gemm_wmma<<<dim3(MT, 6), 256, 0, stream>>>(a16, wFiIn, IN(77),
      nullptr, big, nullptr, M, 768, 256, 0);
  attn_kernel<<<B_ * 4, 256, smF, stream>>>(big, nullptr, o16, 4, 64);
  gemm_wmma<<<dim3(MT, 2), 256, 0, stream>>>(o16, wFiOut, IN(79),
      hbuf, hbuf, nullptr, M, 256, 256, 2);

  // Output head on CLS token.
  ln_kernel<<<B_, 256, 0, stream>>>(hbuf, IN(82), IN(83), cls16, SA_);
  gemm_wmma<<<dim3(B_ / 64, 8), 256, 0, stream>>>(cls16, wFc1, IN(85),
      nullptr, fc1out, nullptr, B_, 1024, 256, 0);
  glu_kernel<<<(B_ * 512) / 256, 256, 0, stream>>>(fc1out, glu16);
  gemm_wmma<<<dim3(B_ / 64, 2), 256, 0, stream>>>(glu16, wFc2, IN(87),
      nullptr, fc2out, nullptr, B_, 256, 512, 0);
  proj_kernel<<<B_, 256, 0, stream>>>(fc2out, IN(88), IN(89), (float*)d_out);
}
